// VectorQuantization_57526791962805
// MI455X (gfx1250) — matrix operand order repaired and checked by reading, not compiled, not run on hardware
//
#include <hip/hip_runtime.h>

// ---- CDNA5 WMMA vector types ----
typedef __attribute__((ext_vector_type(16))) __bf16 v16bf;
typedef __attribute__((ext_vector_type(8)))  __bf16 v8bf;
typedef __attribute__((ext_vector_type(8)))  float  v8f;
typedef __attribute__((ext_vector_type(4)))  float  v4f;

#define DDIM   128
#define KCODES 1024
#define TILES  (KCODES / 16)     // 64 codebook tiles of 16 codes
#define PAD    8                 // bf16 elems of row padding: 272B rows -> conflict-free banks

// ------------------------------------------------------------------
// Kernel 1: c2[k] = sum_d codebook[k][d]^2   (K=1024 rows)
// ------------------------------------------------------------------
__global__ __launch_bounds__(256) void vq_c2_kernel(const float* __restrict__ cb,
                                                    float* __restrict__ c2) {
    int k = blockIdx.x * blockDim.x + threadIdx.x;
    if (k < KCODES) {
        const float* r = cb + (size_t)k * DDIM;
        float s = 0.f;
#pragma unroll
        for (int d = 0; d < DDIM; d += 4) {
            v4f v = *(const v4f*)(r + d);
            s += v.x * v.x + v.y * v.y + v.z * v.z + v.w * v.w;
        }
        c2[k] = s;
    }
}

// ------------------------------------------------------------------
// Kernel 2: argmin_k ( c2[k] - 2 * <z_row, c_k> ) via bf16 hi/lo WMMA
//   256 threads = 8 waves; each wave owns a 16-row tile (128 rows/WG).
// ------------------------------------------------------------------
__global__ __launch_bounds__(256) void vq_argmin_kernel(const float* __restrict__ z,
                                                        const float* __restrict__ cb,
                                                        const float* __restrict__ c2,
                                                        int* __restrict__ out) {
    // Codebook tile staged as pre-split bf16 hi/lo. Row stride 136 bf16 = 272B:
    // lane l reads at byte l*272 + c -> bank 4*l (mod 64): conflict-free b128 reads.
    __shared__ __bf16 s_hi[16][DDIM + PAD];
    __shared__ __bf16 s_lo[16][DDIM + PAD];

    const int tid  = threadIdx.x;
    const int lane = tid & 31;
    const int wave = tid >> 5;
    const int l16  = lane & 15;
    const int lhi  = lane >> 4;              // 0: lanes 0-15, 1: lanes 16-31

    const int rowBase = blockIdx.x * 128 + wave * 16;
    const float* zrow = z + (size_t)(rowBase + l16) * DDIM;

    // ---- Load A fragments (16x32 bf16 per chunk) once into VGPRs ----
    // A layout (ISA 7.12.2, 16-bit A 16x32): lane half selects K sub-block of 8;
    // vector elems 0..7 -> K = base8+e, elems 8..15 -> K = base8+8+e, base8 = 8*lhi.
    v16bf a_hi[4], a_lo[4];
#pragma unroll
    for (int c = 0; c < 4; ++c) {
        const int d0 = c * 32 + lhi * 8;
#pragma unroll
        for (int g = 0; g < 2; ++g) {
            const float* p = zrow + d0 + g * 16;
            v4f x0 = *(const v4f*)(p);
            v4f x1 = *(const v4f*)(p + 4);
#pragma unroll
            for (int e = 0; e < 4; ++e) {
                float xa = x0[e], xb = x1[e];
                __bf16 ha = (__bf16)xa, hb = (__bf16)xb;
                a_hi[c][g * 8 + e]     = ha;
                a_hi[c][g * 8 + 4 + e] = hb;
                a_lo[c][g * 8 + e]     = (__bf16)(xa - (float)ha);
                a_lo[c][g * 8 + 4 + e] = (__bf16)(xb - (float)hb);
            }
        }
    }

    float minv[8];
    int   mini[8];
#pragma unroll
    for (int j = 0; j < 8; ++j) { minv[j] = 3.4e38f; mini[j] = 0; }

    for (int t = 0; t < TILES; ++t) {
        // ---- Stage 16 codebook rows (2048 floats) -> LDS bf16 hi/lo ----
        __syncthreads();                       // previous tile's reads done
        {
            const float* src = cb + (size_t)t * 16 * DDIM;
#pragma unroll
            for (int i = 0; i < 8; ++i) {
                int idx = tid + i * 256;       // coalesced
                float x = src[idx];
                __bf16 h = (__bf16)x;
                int r = idx >> 7, d = idx & 127;
                s_hi[r][d] = h;
                s_lo[r][d] = (__bf16)(x - (float)h);
            }
            if (t + 1 < TILES)                 // global_prefetch_b8 of next tile
                __builtin_prefetch(cb + (size_t)(t + 1) * 16 * DDIM + tid * 8, 0, 1);
        }
        __syncthreads();

        const int  code = t * 16 + l16;
        const float c2v = c2[code];

        // ---- D = A x B, K = 128 in 4 chunks; hi*hi + hi*lo + lo*hi ----
        // B layout (32x16 bf16): lane = N (code), elems 0..7 -> K = 8*lhi + e, elems 8..15 -> K = 16 + 8*lhi + (e-8).
        v8f acc = {};
#pragma unroll
        for (int c = 0; c < 4; ++c) {
            const int db = c * 32 + lhi * 8;
            union { v16bf v; v8bf h[2]; } bh, bl;   // 16B-aligned LDS halves
            bh.h[0] = *(const v8bf*)&s_hi[l16][db];
            bh.h[1] = *(const v8bf*)&s_hi[l16][db + 16];
            bl.h[0] = *(const v8bf*)&s_lo[l16][db];
            bl.h[1] = *(const v8bf*)&s_lo[l16][db + 16];
            acc = __builtin_amdgcn_wmma_f32_16x16x32_bf16(false, a_hi[c], false, bh.v,
                                                          (short)0, acc, false, false);
            acc = __builtin_amdgcn_wmma_f32_16x16x32_bf16(false, a_hi[c], false, bl.v,
                                                          (short)0, acc, false, false);
            acc = __builtin_amdgcn_wmma_f32_16x16x32_bf16(false, a_lo[c], false, bh.v,
                                                          (short)0, acc, false, false);
        }

        // C/D layout: VGPR j, lanes 0-15 -> row j, lanes 16-31 -> row j+8; N = l16.
#pragma unroll
        for (int j = 0; j < 8; ++j) {
            float score = c2v - 2.0f * acc[j];
            if (score < minv[j]) { minv[j] = score; mini[j] = code; }
        }
    }

    // ---- Reduce (min, idx) across the 16 lanes holding each row ----
#pragma unroll
    for (int j = 0; j < 8; ++j) {
        float v = minv[j];
        int   i = mini[j];
#pragma unroll
        for (int m = 8; m >= 1; m >>= 1) {     // masks 1..8 stay in each half-wave
            float ov = __shfl_xor(v, m, 32);
            int   oi = __shfl_xor(i, m, 32);
            if (ov < v || (ov == v && oi < i)) { v = ov; i = oi; }
        }
        if (l16 == 0) out[rowBase + lhi * 8 + j] = i;
    }
}

// ------------------------------------------------------------------
extern "C" void kernel_launch(void* const* d_in, const int* in_sizes, int n_in,
                              void* d_out, int out_size, void* d_ws, size_t ws_size,
                              hipStream_t stream) {
    (void)in_sizes; (void)n_in; (void)out_size; (void)ws_size;
    const float* z_e = (const float*)d_in[0];   // (64,32,32,128) f32
    const float* cb  = (const float*)d_in[1];   // (1024,128)     f32
    float* c2  = (float*)d_ws;                  // 1024 floats scratch
    int*   out = (int*)d_out;                   // 65536 int32 indices

    hipLaunchKernelGGL(vq_c2_kernel, dim3(4), dim3(256), 0, stream, cb, c2);
    hipLaunchKernelGGL(vq_argmin_kernel, dim3(65536 / 128), dim3(256), 0, stream,
                       z_e, cb, c2, out);
}
